// SimpleDecoderLayer_80719615361304
// MI455X (gfx1250) — matrix operand order repaired and checked by reading, not compiled, not run on hardware
//
#include <hip/hip_runtime.h>
#include <math.h>

// ---------------- problem constants ----------------
#define BATCH   2
#define SEQ     2048
#define DM      2048
#define NHEAD   16
#define HDIM    128
#define FF      8192
#define TOK     (BATCH * SEQ)     // 4096 rows
#define LNEPS   1e-8f

// ---------------- types ----------------
typedef __attribute__((ext_vector_type(16))) __bf16 v16bf;
typedef __attribute__((ext_vector_type(8)))  float  v8f;

// RNE float -> bf16 (bit-level)
__device__ __forceinline__ __bf16 f2bf(float f) {
    union { float f; unsigned u; } a; a.f = f;
    unsigned r = a.u + 0x7FFFu + ((a.u >> 16) & 1u);
    union { unsigned short s; __bf16 b; } o; o.s = (unsigned short)(r >> 16);
    return o.b;
}

// load a 16-element bf16 fragment as two 16B chunks
__device__ __forceinline__ v16bf load2(const __bf16* p0, const __bf16* p1) {
    union { uint4 u[2]; v16bf v; } t;
    t.u[0] = *reinterpret_cast<const uint4*>(p0);
    t.u[1] = *reinterpret_cast<const uint4*>(p1);
    return t.v;
}

__device__ __forceinline__ v8f wmma_bf16(v16bf a, v16bf b, v8f c) {
    return __builtin_amdgcn_wmma_f32_16x16x32_bf16(
        false, a, false, b, (short)0, c, false, false);
}

__device__ __forceinline__ float gelu_exact(float x) {
    return 0.5f * x * (1.0f + erff(x * 0.70710678118654752f));
}

// CDNA5 async global->LDS copy, 16B per lane, tracked by ASYNCcnt.
// LDS dest VGPR = low 32 bits of the generic shared-memory address
// (generic LDS addresses carry the LDS byte offset in addr[31:0]).
__device__ __forceinline__ void async_copy_b128(const __bf16* gsrc, __bf16* ldst) {
    unsigned lds_off = (unsigned)(unsigned long long)ldst;
    unsigned long long ga = (unsigned long long)gsrc;
    asm volatile("global_load_async_to_lds_b128 %0, %1, off"
                 :: "v"(lds_off), "v"(ga) : "memory");
}
__device__ __forceinline__ void wait_async0() {
    asm volatile("s_wait_asynccnt 0x0" ::: "memory");
}

// ---------------- weight transpose+cast: W[K][N] f32 -> Wt[N][K] bf16 ----------------
__global__ __launch_bounds__(256) void transpose_cast_kernel(
    const float* __restrict__ W, __bf16* __restrict__ Wt, int K, int N) {
    __shared__ float t[32][33];
    int tx = threadIdx.x & 31;
    int ty = threadIdx.x >> 5;           // 0..7
    int n0 = blockIdx.x * 32;
    int k0 = blockIdx.y * 32;
#pragma unroll
    for (int j = 0; j < 4; ++j)
        t[ty + j * 8][tx] = W[(size_t)(k0 + ty + j * 8) * N + n0 + tx];
    __syncthreads();
#pragma unroll
    for (int j = 0; j < 4; ++j)
        Wt[(size_t)(n0 + ty + j * 8) * K + k0 + tx] = f2bf(t[tx][ty + j * 8]);
}

// ---------------- GEMM: C[M,N] = A[M,K] * Bt[N,K]^T + bias, optional GELU ----------------
// 128x128 block tile, BK=32, 256 threads = 8 waves (4m x 2n).
// Double-buffered LDS; B tile (and bf16 A tile) filled by async-to-LDS DMA
// overlapping WMMA on the other buffer; f32 A tile is register-staged
// (needs f32->bf16 conversion on the way in).
template <bool A_BF16, int ACT, bool OUT_BF16>
__global__ __launch_bounds__(256) void gemm_kernel(
    const void* __restrict__ Av, const __bf16* __restrict__ Bt,
    const float* __restrict__ bias, void* __restrict__ Cv,
    int M, int N, int K) {
    __shared__ __bf16 As[2][128][40];
    __shared__ __bf16 Bs[2][128][40];

    const int tid  = threadIdx.x;
    const int lane = tid & 31;
    const int wid  = tid >> 5;
    const int wm   = wid & 3;       // 0..3 -> 32 rows each
    const int wn   = wid >> 2;      // 0..1 -> 64 cols each
    const int half = lane >> 4;
    const int nn   = lane & 15;
    const int row0 = blockIdx.y * 128;
    const int col0 = blockIdx.x * 128;

    v8f acc[2][4];
#pragma unroll
    for (int a = 0; a < 2; ++a)
#pragma unroll
        for (int b = 0; b < 4; ++b)
#pragma unroll
            for (int i = 0; i < 8; ++i) acc[a][b][i] = 0.0f;

    // per-thread tile-fill coordinates
    const int br0 = (tid) >> 2,        bc0 = (tid & 3) * 8;          // uint4 slots
    const int br1 = (tid + 256) >> 2,  bc1 = ((tid + 256) & 3) * 8;

    auto issueB = [&](int buf, int kt) {
        async_copy_b128(Bt + (size_t)(col0 + br0) * K + kt + bc0, &Bs[buf][br0][bc0]);
        async_copy_b128(Bt + (size_t)(col0 + br1) * K + kt + bc1, &Bs[buf][br1][bc1]);
    };
    auto issueA_bf16 = [&](int buf, int kt) {
        const __bf16* A = (const __bf16*)Av;
        async_copy_b128(A + (size_t)(row0 + br0) * K + kt + bc0, &As[buf][br0][bc0]);
        async_copy_b128(A + (size_t)(row0 + br1) * K + kt + bc1, &As[buf][br1][bc1]);
    };

    float4 areg[4];
    auto fetchA_f32 = [&](int kt) {
        const float* A = (const float*)Av;
#pragma unroll
        for (int i = 0; i < 4; ++i) {
            int idx = tid + i * 256;              // 1024 float4 = 128x32
            int r = idx >> 3, c4 = idx & 7;
            areg[i] = *reinterpret_cast<const float4*>(
                A + (size_t)(row0 + r) * K + kt + c4 * 4);
        }
    };
    auto storeA_f32 = [&](int buf) {
#pragma unroll
        for (int i = 0; i < 4; ++i) {
            int idx = tid + i * 256;
            int r = idx >> 3, c4 = idx & 7;
            union { __bf16 b[4]; uint2 u; } pk;
            pk.b[0] = f2bf(areg[i].x); pk.b[1] = f2bf(areg[i].y);
            pk.b[2] = f2bf(areg[i].z); pk.b[3] = f2bf(areg[i].w);
            *reinterpret_cast<uint2*>(&As[buf][r][c4 * 4]) = pk.u;
        }
    };

    // ---- prologue: fill buffer 0 ----
    issueB(0, 0);
    if constexpr (A_BF16) issueA_bf16(0, 0);
    else { fetchA_f32(0); storeA_f32(0); }
    wait_async0();
    __syncthreads();

    const int nk = K >> 5;
    for (int t = 0; t < nk; ++t) {
        const int cur = t & 1, nxt = cur ^ 1;
        const int ktn = (t + 1) << 5;
        const bool more = (t + 1 < nk);
        if (more) {
            issueB(nxt, ktn);
            if constexpr (A_BF16) issueA_bf16(nxt, ktn);
            else fetchA_f32(ktn);
        }

        v16bf af[2], bf[4];
#pragma unroll
        for (int tm = 0; tm < 2; ++tm) {
            int r = wm * 32 + tm * 16 + nn;
            af[tm] = load2(&As[cur][r][half * 8], &As[cur][r][16 + half * 8]);
        }
#pragma unroll
        for (int tn = 0; tn < 4; ++tn) {
            int r = wn * 64 + tn * 16 + nn;
            bf[tn] = load2(&Bs[cur][r][half * 8], &Bs[cur][r][16 + half * 8]);
        }
#pragma unroll
        for (int tm = 0; tm < 2; ++tm)
#pragma unroll
            for (int tn = 0; tn < 4; ++tn)
                acc[tm][tn] = wmma_bf16(af[tm], bf[tn], acc[tm][tn]);

        if (more) {
            if constexpr (!A_BF16) storeA_f32(nxt);
            wait_async0();
        }
        __syncthreads();
    }

    // epilogue: bias (+GELU) -> f32 or bf16
#pragma unroll
    for (int tm = 0; tm < 2; ++tm)
#pragma unroll
        for (int tn = 0; tn < 4; ++tn)
#pragma unroll
            for (int i = 0; i < 8; ++i) {
                int gr = row0 + wm * 32 + tm * 16 + half * 8 + i;
                int gc = col0 + wn * 64 + tn * 16 + nn;
                float v = acc[tm][tn][i] + bias[gc];
                if constexpr (ACT == 1) v = gelu_exact(v);
                if constexpr (OUT_BF16)
                    ((__bf16*)Cv)[(size_t)gr * N + gc] = f2bf(v);
                else
                    ((float*)Cv)[(size_t)gr * N + gc] = v;
            }
}

// ---------------- repack projections for attention ----------------
__global__ __launch_bounds__(256) void repack_bhsd(
    const float* __restrict__ src, __bf16* __restrict__ dst) {
    size_t idx = (size_t)blockIdx.x * 256 + threadIdx.x;
    int d = idx & (HDIM - 1);
    int h = (idx >> 7) & (NHEAD - 1);
    int s = (idx >> 11) & (SEQ - 1);
    int b = (int)(idx >> 22);
    dst[(((size_t)(b * NHEAD + h) * SEQ + s) * HDIM) + d] = f2bf(src[idx]);
}
__global__ __launch_bounds__(256) void repack_bhds(
    const float* __restrict__ src, __bf16* __restrict__ dst) {
    size_t idx = (size_t)blockIdx.x * 256 + threadIdx.x;
    int d = idx & (HDIM - 1);
    int h = (idx >> 7) & (NHEAD - 1);
    int s = (idx >> 11) & (SEQ - 1);
    int b = (int)(idx >> 22);
    dst[(((size_t)(b * NHEAD + h) * HDIM + d) * SEQ) + s] = f2bf(src[idx]);
}

// ---------------- causal flash attention ----------------
// grid: (SEQ/128, BATCH*NHEAD), 256 threads; wave w owns q rows [qb0+16w, +16)
__global__ __launch_bounds__(256) void attn_kernel(
    const __bf16* __restrict__ Qg, const __bf16* __restrict__ Kg,
    const __bf16* __restrict__ VTg, float* __restrict__ Og) {
    __shared__ __bf16 Pl[8][16][32];   // per-wave P staging (C-frag -> A-frag)

    const int tid  = threadIdx.x;
    const int lane = tid & 31;
    const int w    = tid >> 5;
    const int half = lane >> 4;
    const int nn   = lane & 15;
    const int bh   = blockIdx.y;
    const int b    = bh >> 4;
    const int h    = bh & 15;
    const int qb0  = blockIdx.x * 128;
    const int rbase = qb0 + w * 16;

    const __bf16* Q  = Qg  + (size_t)bh * SEQ * HDIM;
    const __bf16* Kp = Kg  + (size_t)bh * SEQ * HDIM;
    const __bf16* VT = VTg + (size_t)bh * HDIM * SEQ;

    v16bf aq[4];
    {
        const __bf16* qrow = Q + (size_t)(rbase + nn) * HDIM;
#pragma unroll
        for (int c = 0; c < 4; ++c)
            aq[c] = load2(qrow + c * 32 + half * 8, qrow + c * 32 + 16 + half * 8);
    }

    v8f o[8];
    float m[8], l[8];
#pragma unroll
    for (int t = 0; t < 8; ++t)
#pragma unroll
        for (int i = 0; i < 8; ++i) o[t][i] = 0.0f;
#pragma unroll
    for (int i = 0; i < 8; ++i) { m[i] = -3.4e38f; l[i] = 0.0f; }

    const float scale = 0.08838834764831845f;   // 1/sqrt(128)
    const int rend = rbase + 16;                // causal edge for this wave

    for (int kb = 0; kb < rend; kb += 32) {
        // prefetch next key block's K rows and V columns into cache
        if (kb + 32 < rend) {
            __builtin_prefetch(Kp + (size_t)(kb + 32 + lane) * HDIM, 0, 3);
#pragma unroll
            for (int t = 0; t < 4; ++t)
                __builtin_prefetch(VT + (size_t)(t * 32 + lane) * SEQ + kb + 32, 0, 3);
        }

        v8f s0, s1;
#pragma unroll
        for (int i = 0; i < 8; ++i) { s0[i] = 0.0f; s1[i] = 0.0f; }
#pragma unroll
        for (int c = 0; c < 4; ++c) {
            const __bf16* k0 = Kp + (size_t)(kb + nn) * HDIM + c * 32 + half * 8;
            s0 = wmma_bf16(aq[c], load2(k0, k0 + 16), s0);
            const __bf16* k1 = Kp + (size_t)(kb + 16 + nn) * HDIM + c * 32 + half * 8;
            s1 = wmma_bf16(aq[c], load2(k1, k1 + 16), s1);
        }

        float p0[8], p1[8], alpha[8];
#pragma unroll
        for (int i = 0; i < 8; ++i) {
            int qr = rbase + half * 8 + i;
            float x0 = s0[i] * scale; if (kb + nn > qr)      x0 = -1e9f;
            float x1 = s1[i] * scale; if (kb + 16 + nn > qr) x1 = -1e9f;
            float mt = fmaxf(x0, x1);
            mt = fmaxf(mt, __shfl_xor(mt, 1, 32));
            mt = fmaxf(mt, __shfl_xor(mt, 2, 32));
            mt = fmaxf(mt, __shfl_xor(mt, 4, 32));
            mt = fmaxf(mt, __shfl_xor(mt, 8, 32));
            float mn = fmaxf(m[i], mt);
            float a  = __expf(m[i] - mn);
            float e0 = __expf(x0 - mn), e1 = __expf(x1 - mn);
            float rs = e0 + e1;
            rs += __shfl_xor(rs, 1, 32);
            rs += __shfl_xor(rs, 2, 32);
            rs += __shfl_xor(rs, 4, 32);
            rs += __shfl_xor(rs, 8, 32);
            l[i] = l[i] * a + rs;
            m[i] = mn;
            alpha[i] = a; p0[i] = e0; p1[i] = e1;
        }
#pragma unroll
        for (int t = 0; t < 8; ++t)
#pragma unroll
            for (int i = 0; i < 8; ++i) o[t][i] *= alpha[i];

        // stage P tile (16x32) through per-wave LDS: C-frag -> A-frag
#pragma unroll
        for (int i = 0; i < 8; ++i) {
            Pl[w][half * 8 + i][nn]      = f2bf(p0[i]);
            Pl[w][half * 8 + i][16 + nn] = f2bf(p1[i]);
        }
        v16bf ap = load2(&Pl[w][nn][half * 8], &Pl[w][nn][16 + half * 8]);

#pragma unroll
        for (int t = 0; t < 8; ++t) {
            const __bf16* vrow = VT + (size_t)(t * 16 + nn) * SEQ + kb + half * 8;
            o[t] = wmma_bf16(ap, load2(vrow, vrow + 16), o[t]);
        }
    }

#pragma unroll
    for (int t = 0; t < 8; ++t)
#pragma unroll
        for (int i = 0; i < 8; ++i) {
            int sr = rbase + half * 8 + i;
            int d  = t * 16 + nn;
            Og[((size_t)(b * SEQ + sr)) * DM + h * HDIM + d] = o[t][i] / l[i];
        }
}

// ---------------- residual + LayerNorm (one row per block) ----------------
__global__ __launch_bounds__(256) void add_ln_kernel(
    const float* __restrict__ X, const float* __restrict__ R,
    const float* __restrict__ g, const float* __restrict__ be,
    float* __restrict__ Y) {
    int row = blockIdx.x, tid = threadIdx.x;
    const float* x = X + (size_t)row * DM;
    const float* r = R + (size_t)row * DM;
    float v[8], s = 0.0f, s2 = 0.0f;
#pragma unroll
    for (int j = 0; j < 8; ++j) {
        int c = tid + j * 256;
        float t = x[c] + r[c];
        v[j] = t; s += t; s2 += t * t;
    }
#pragma unroll
    for (int msk = 1; msk < 32; msk <<= 1) {
        s  += __shfl_xor(s,  msk, 32);
        s2 += __shfl_xor(s2, msk, 32);
    }
    __shared__ float rs[8], rs2[8];
    if ((tid & 31) == 0) { rs[tid >> 5] = s; rs2[tid >> 5] = s2; }
    __syncthreads();
    float ts = 0.0f, ts2 = 0.0f;
#pragma unroll
    for (int j = 0; j < 8; ++j) { ts += rs[j]; ts2 += rs2[j]; }
    float mean = ts / (float)DM;
    float var  = ts2 / (float)DM - mean * mean;
    float rstd = rsqrtf(var + LNEPS);
    float* y = Y + (size_t)row * DM;
#pragma unroll
    for (int j = 0; j < 8; ++j) {
        int c = tid + j * 256;
        y[c] = (v[j] - mean) * rstd * g[c] + be[c];
    }
}

// ---------------- host orchestration ----------------
extern "C" void kernel_launch(void* const* d_in, const int* in_sizes, int n_in,
                              void* d_out, int out_size, void* d_ws, size_t ws_size,
                              hipStream_t stream) {
    const float* x  = (const float*)d_in[0];
    const float* wq = (const float*)d_in[1];  const float* bq = (const float*)d_in[2];
    const float* wk = (const float*)d_in[3];  const float* bk = (const float*)d_in[4];
    const float* wv = (const float*)d_in[5];  const float* bv = (const float*)d_in[6];
    const float* wo = (const float*)d_in[7];  const float* bo = (const float*)d_in[8];
    const float* g1 = (const float*)d_in[9];  const float* b1 = (const float*)d_in[10];
    const float* wu = (const float*)d_in[11]; const float* bu = (const float*)d_in[12];
    const float* wd = (const float*)d_in[13]; const float* bd = (const float*)d_in[14];
    const float* g2 = (const float*)d_in[15]; const float* b2 = (const float*)d_in[16];

    char* ws = (char*)d_ws;
    constexpr size_t MiB = 1ull << 20;
    __bf16* wqT = (__bf16*)(ws + 0);          // D*D bf16 = 8 MiB each
    __bf16* wkT = (__bf16*)(ws + 8 * MiB);
    __bf16* wvT = (__bf16*)(ws + 16 * MiB);
    __bf16* woT = (__bf16*)(ws + 24 * MiB);
    __bf16* wuT = (__bf16*)(ws + 32 * MiB);   // F*D bf16 = 32 MiB
    __bf16* wdT = (__bf16*)(ws + 64 * MiB);   // D*F bf16 = 32 MiB
    float*  qf  = (float*)(ws + 96 * MiB);    // TOK*D f32 = 32 MiB (later: attn out)
    float*  kf  = (float*)(ws + 128 * MiB);   // (later: o-proj out)
    float*  vf  = (float*)(ws + 160 * MiB);   // (later: down-proj out)
    __bf16* qb  = (__bf16*)(ws + 192 * MiB);  // [B,H,S,HD] bf16 = 16 MiB
    __bf16* kb  = (__bf16*)(ws + 208 * MiB);
    __bf16* vT  = (__bf16*)(ws + 224 * MiB);  // [B,H,HD,S]
    float*  ln1 = (float*)(ws + 240 * MiB);   // 32 MiB
    __bf16* hb  = (__bf16*)(ws + 272 * MiB);  // TOK*F bf16 = 64 MiB
    float*  ao  = qf;
    float*  po  = kf;
    float*  df  = vf;

    const dim3 blk(256);

    transpose_cast_kernel<<<dim3(DM / 32, DM / 32), blk, 0, stream>>>(wq, wqT, DM, DM);
    transpose_cast_kernel<<<dim3(DM / 32, DM / 32), blk, 0, stream>>>(wk, wkT, DM, DM);
    transpose_cast_kernel<<<dim3(DM / 32, DM / 32), blk, 0, stream>>>(wv, wvT, DM, DM);
    transpose_cast_kernel<<<dim3(DM / 32, DM / 32), blk, 0, stream>>>(wo, woT, DM, DM);
    transpose_cast_kernel<<<dim3(FF / 32, DM / 32), blk, 0, stream>>>(wu, wuT, DM, FF);
    transpose_cast_kernel<<<dim3(DM / 32, FF / 32), blk, 0, stream>>>(wd, wdT, FF, DM);

    gemm_kernel<false, 0, false><<<dim3(DM / 128, TOK / 128), blk, 0, stream>>>(
        x, wqT, bq, qf, TOK, DM, DM);
    gemm_kernel<false, 0, false><<<dim3(DM / 128, TOK / 128), blk, 0, stream>>>(
        x, wkT, bk, kf, TOK, DM, DM);
    gemm_kernel<false, 0, false><<<dim3(DM / 128, TOK / 128), blk, 0, stream>>>(
        x, wvT, bv, vf, TOK, DM, DM);

    const int nrep = (TOK * DM) / 256;
    repack_bhsd<<<nrep, blk, 0, stream>>>(qf, qb);
    repack_bhsd<<<nrep, blk, 0, stream>>>(kf, kb);
    repack_bhds<<<nrep, blk, 0, stream>>>(vf, vT);

    attn_kernel<<<dim3(SEQ / 128, BATCH * NHEAD), blk, 0, stream>>>(qb, kb, vT, ao);

    gemm_kernel<false, 0, false><<<dim3(DM / 128, TOK / 128), blk, 0, stream>>>(
        ao, woT, bo, po, TOK, DM, DM);

    add_ln_kernel<<<TOK, blk, 0, stream>>>(x, po, g1, b1, ln1);

    gemm_kernel<false, 1, true><<<dim3(FF / 128, TOK / 128), blk, 0, stream>>>(
        ln1, wuT, bu, hb, TOK, FF, DM);

    gemm_kernel<true, 0, false><<<dim3(DM / 128, TOK / 128), blk, 0, stream>>>(
        hb, wdT, bd, df, TOK, DM, FF);

    add_ln_kernel<<<TOK, blk, 0, stream>>>(ln1, df, g2, b2, (float*)d_out);
}
